// UnaryLinearComplex_54949811585181
// MI455X (gfx1250) — compile-verified, matrix operand order checked by reading, not run on hardware
//
#include <hip/hip_runtime.h>

typedef __attribute__((ext_vector_type(16))) _Float16 v16h;
typedef __attribute__((ext_vector_type(8)))  float    v8f;

#define IN_DIM  2048
#define OUT_DIM 2048
#define THRESH  2047.5f   // OFFSET = (2*IN-1)/2

#define LD8F(dst, ptr) do { \
    const float4* _p = reinterpret_cast<const float4*>(ptr); \
    float4 _a = _p[0]; float4 _b = _p[1]; \
    dst[0]=_a.x; dst[1]=_a.y; dst[2]=_a.z; dst[3]=_a.w; \
    dst[4]=_b.x; dst[5]=_b.y; dst[6]=_b.z; dst[7]=_b.w; } while(0)

#define LD8I(dst, ptr) do { \
    const int4* _p = reinterpret_cast<const int4*>(ptr); \
    int4 _a = _p[0]; int4 _b = _p[1]; \
    dst[0]=_a.x; dst[1]=_a.y; dst[2]=_a.z; dst[3]=_a.w; \
    dst[4]=_b.x; dst[5]=_b.y; dst[6]=_b.z; dst[7]=_b.w; } while(0)

__global__ void ulc_zero(float* ws) {
    int j = blockIdx.x * blockDim.x + threadIdx.x;
    if (j < 2 * OUT_DIM) ws[j] = 0.0f;
}

// Grid: (OUT/16, 4). Block: 256 threads (8 waves).
// Wave w of column-split cy covers columns [cy*512 + w*64, +64) for one 16-row tile.
// Per 32-column chunk we build two f16 A tiles (real/imag contribution values) and
// row-sum them via v_wmma_f32_16x16x32_f16 against an all-ones B matrix.
__global__ __launch_bounds__(256) void ulc_main(
    const float* __restrict__ input_r, const float* __restrict__ input_i,
    const float* __restrict__ src_r,   const float* __restrict__ src_i,
    const float* __restrict__ rng,
    const int* __restrict__ i1rr, const int* __restrict__ i0rr,   // idx1/idx0 fr_wr
    const int* __restrict__ i1ri, const int* __restrict__ i0ri,   // fr_wi
    const int* __restrict__ i1ir, const int* __restrict__ i0ir,   // fi_wr
    const int* __restrict__ i1ii, const int* __restrict__ i0ii,   // fi_wi
    float* __restrict__ ws)
{
    __shared__ float rngL[256];
    const int tid = threadIdx.x;
    rngL[tid] = rng[tid];            // blockDim.x == 256 == RNG length
    __syncthreads();

    const int lane    = tid & 31;
    const int wave    = tid >> 5;
    const int m       = lane & 15;   // A/D row within tile
    const int hi      = lane >> 4;   // half-wave selects K sub-block / D row group
    const int rowBase = blockIdx.x * 16;
    const int rowOff  = (rowBase + m) * IN_DIM;
    const int cs      = blockIdx.y * (IN_DIM / 4);   // 512-column split

    v16h ones;
    #pragma unroll
    for (int j = 0; j < 16; ++j) ones[j] = (_Float16)1.0f;

    v8f accR = {};   // accumulates o_fr_wr + o_fi_wi  (-> out_r)
    v8f accI = {};   // accumulates o_fr_wi + o_fi_wr  (-> out_i)

    for (int s = 0; s < 2; ++s) {
        const int kBase = cs + wave * 64 + s * 32;
        v16h aR, aI;
        #pragma unroll
        for (int half = 0; half < 2; ++half) {
            const int c0 = kBase + hi * 8 + half * 16;   // 8 consecutive columns
            float sr[8], si[8], xr[8], xi[8];
            int a1rr[8], a0rr[8], a1ri[8], a0ri[8];
            int a1ir[8], a0ir[8], a1ii[8], a0ii[8];
            LD8F(sr, src_r + rowOff + c0);
            LD8F(si, src_i + rowOff + c0);
            LD8F(xr, input_r + c0);
            LD8F(xi, input_i + c0);
            LD8I(a1rr, i1rr + rowOff + c0);  LD8I(a0rr, i0rr + rowOff + c0);
            LD8I(a1ri, i1ri + rowOff + c0);  LD8I(a0ri, i0ri + rowOff + c0);
            LD8I(a1ir, i1ir + rowOff + c0);  LD8I(a0ir, i0ir + rowOff + c0);
            LD8I(a1ii, i1ii + rowOff + c0);  LD8I(a0ii, i0ii + rowOff + c0);
            #pragma unroll
            for (int j = 0; j < 8; ++j) {
                const float b1rr = (sr[j] > rngL[a1rr[j] & 255]) ? 1.0f : 0.0f;
                const float b0rr = (sr[j] > rngL[a0rr[j] & 255]) ? 1.0f : 0.0f;
                const float b1ri = (si[j] > rngL[a1ri[j] & 255]) ? 1.0f : 0.0f;
                const float b0ri = (si[j] > rngL[a0ri[j] & 255]) ? 1.0f : 0.0f;
                const float b1ir = (sr[j] > rngL[a1ir[j] & 255]) ? 1.0f : 0.0f;
                const float b0ir = (sr[j] > rngL[a0ir[j] & 255]) ? 1.0f : 0.0f;
                const float b1ii = (si[j] > rngL[a1ii[j] & 255]) ? 1.0f : 0.0f;
                const float b0ii = (si[j] > rngL[a0ii[j] & 255]) ? 1.0f : 0.0f;
                const float x  = xr[j], y = xi[j];
                // o_fr_wr (x) + o_fi_wi (1-y)   -> real accumulator
                const float cr = x * b1rr + (1.0f - x) * (1.0f - b0rr)
                               + (1.0f - y) * b1ii + y * (1.0f - b0ii);
                // o_fr_wi (x) + o_fi_wr (y)     -> imag accumulator
                const float ci = x * b1ri + (1.0f - x) * (1.0f - b0ri)
                               + y * b1ir + (1.0f - y) * (1.0f - b0ir);
                aR[half * 8 + j] = (_Float16)cr;
                aI[half * 8 + j] = (_Float16)ci;
            }
        }
        accR = __builtin_amdgcn_wmma_f32_16x16x32_f16(
            false, aR, false, ones, (short)0, accR, false, false);
        accI = __builtin_amdgcn_wmma_f32_16x16x32_f16(
            false, aI, false, ones, (short)0, accI, false, false);
    }

    // D layout: VGPR r <-> row r (lanes 0-15) / row r+8 (lanes 16-31); all N equal.
    if (m == 0) {
        const int rb = rowBase + hi * 8;
        #pragma unroll
        for (int r = 0; r < 8; ++r) {
            atomicAdd(&ws[rb + r],           accR[r]);
            atomicAdd(&ws[OUT_DIM + rb + r], accI[r]);
        }
    }
}

__global__ void ulc_finalize(const float* __restrict__ ws, float* __restrict__ out) {
    int j = blockIdx.x * blockDim.x + threadIdx.x;
    if (j < 2 * OUT_DIM) out[j] = (ws[j] > THRESH) ? 1.0f : 0.0f;
}

extern "C" void kernel_launch(void* const* d_in, const int* in_sizes, int n_in,
                              void* d_out, int out_size, void* d_ws, size_t ws_size,
                              hipStream_t stream) {
    (void)in_sizes; (void)n_in; (void)out_size; (void)ws_size;
    const float* input_r = (const float*)d_in[0];
    const float* input_i = (const float*)d_in[1];
    const float* src_r   = (const float*)d_in[2];
    const float* src_i   = (const float*)d_in[3];
    const float* rng     = (const float*)d_in[4];
    const int*   i1rr    = (const int*)d_in[5];
    const int*   i0rr    = (const int*)d_in[6];
    const int*   i1ri    = (const int*)d_in[7];
    const int*   i0ri    = (const int*)d_in[8];
    const int*   i1ir    = (const int*)d_in[9];
    const int*   i0ir    = (const int*)d_in[10];
    const int*   i1ii    = (const int*)d_in[11];
    const int*   i0ii    = (const int*)d_in[12];
    float* ws  = (float*)d_ws;
    float* out = (float*)d_out;

    ulc_zero<<<dim3((2 * OUT_DIM + 255) / 256), 256, 0, stream>>>(ws);
    ulc_main<<<dim3(OUT_DIM / 16, 4), 256, 0, stream>>>(
        input_r, input_i, src_r, src_i, rng,
        i1rr, i0rr, i1ri, i0ri, i1ir, i0ir, i1ii, i0ii, ws);
    ulc_finalize<<<dim3((2 * OUT_DIM + 255) / 256), 256, 0, stream>>>(ws, out);
}
